// StyledConv_13125420056757
// MI455X (gfx1250) — matrix operand order repaired and checked by reading, not compiled, not run on hardware
//
#include <hip/hip_runtime.h>
#include <math.h>

// ---------------- CDNA5 WMMA types ----------------
typedef __attribute__((ext_vector_type(16))) __bf16 v16bf;
typedef __attribute__((ext_vector_type(8)))  float  v8f;

union FragBF { unsigned u[8]; v16bf v; };

// Async global->LDS staging (gfx1250 GLOBAL_LOAD_ASYNC_TO_LDS_B128, ASYNCcnt-tracked)
// Probe-derived signature: (v4i addrspace(1)*, v4i addrspace(3)*, imm int, imm int cpol)
#if defined(__gfx1250__) && __has_builtin(__builtin_amdgcn_global_load_async_to_lds_b128) && \
    __has_builtin(__builtin_amdgcn_s_wait_asynccnt)
#define USE_ASYNC_LDS 1
typedef int v4i_vec __attribute__((vector_size(16)));
typedef __attribute__((address_space(1))) v4i_vec* gv4i_p;
typedef __attribute__((address_space(3))) v4i_vec* lv4i_p;
#else
#define USE_ASYNC_LDS 0
#endif

__device__ __forceinline__ unsigned short f32_to_bf16(float f) {
    unsigned b = __float_as_uint(f);
    unsigned r = b + 0x7FFFu + ((b >> 16) & 1u);   // round to nearest even
    return (unsigned short)(r >> 16);
}

// ---------------- 1) InstanceNorm statistics ----------------
__global__ void __launch_bounds__(256)
instnorm_stats(const float* __restrict__ x, float* __restrict__ mu,
               float* __restrict__ rs, int HW)
{
    __shared__ float s1[256];
    __shared__ float s2[256];
    const int bc = blockIdx.x;
    const float* p = x + (size_t)bc * HW;
    float a = 0.f, b = 0.f;
    for (int i = threadIdx.x; i < HW; i += 256) { float v = p[i]; a += v; b += v * v; }
    s1[threadIdx.x] = a; s2[threadIdx.x] = b;
    __syncthreads();
    for (int off = 128; off > 0; off >>= 1) {
        if ((int)threadIdx.x < off) {
            s1[threadIdx.x] += s1[threadIdx.x + off];
            s2[threadIdx.x] += s2[threadIdx.x + off];
        }
        __syncthreads();
    }
    if (threadIdx.x == 0) {
        float m   = s1[0] / (float)HW;
        float var = s2[0] / (float)HW - m * m;
        mu[bc] = m;
        rs[bc] = rsqrtf(var + 1e-5f);
    }
}

// ---------------- 2) Weight pack: [Co][Ci][3][3] f32 -> [tap][Co][Ci] bf16*scale ----------------
__global__ void pack_weight(const float* __restrict__ w, unsigned short* __restrict__ wp,
                            float scale, int Cout, int Cin, int total)
{
    int idx = blockIdx.x * blockDim.x + threadIdx.x;
    if (idx >= total) return;
    int t   = idx % 9;
    int rem = idx / 9;
    int ci  = rem % Cin;
    int co  = rem / Cin;
    wp[((size_t)t * Cout + co) * Cin + ci] = f32_to_bf16(w[idx] * scale);
}

// ---------------- 3) Activation pack: NCHW f32 -> NHWC bf16 ----------------
__global__ void pack_nhwc(const float* __restrict__ src, unsigned short* __restrict__ dst,
                          int C, int HW, int total)
{
    int idx = blockIdx.x * blockDim.x + threadIdx.x;
    if (idx >= total) return;
    int p  = idx % HW;
    int bc = idx / HW;
    int c  = bc % C;
    int b  = bc / C;
    dst[((size_t)b * HW + p) * C + c] = f32_to_bf16(src[idx]);
}

// ---------------- epilogue helper ----------------
__device__ __forceinline__ void epilog_tile(const v8f& acc, bool nv, int aO, int bO,
                                            int mode, int py, int px,
                                            float* obase, int OutH, int OutW,
                                            int coBase, int half, const float* bias)
{
    if (!nv) return;
    const int oy = (mode == 0) ? aO : (2 * aO + py);
    const int ox = (mode == 0) ? bO : (2 * bO + px);
    float* ob = obase + (size_t)oy * OutW + ox;
#pragma unroll
    for (int i = 0; i < 8; ++i) {
        const int co = coBase + i + 8 * half;
        float v = acc[i];
        if (bias) v += bias[co];
        ob[(size_t)co * OutH * OutW] = v;
    }
}

// ---------------- 4) Implicit-GEMM conv via v_wmma_f32_16x16x32_bf16 ----------------
// Workgroup: 128 threads = 4 waves arranged 2(M)x2(N); macro-tile 64(Cout) x 64(spatial).
// Each wave: 2x2 of 16x16 WMMA tiles (4 f32 accumulators).
// LDS staging: per (tap,k0) step, A tile 64x32 bf16 and B tile 64x32 bf16 (K contiguous),
// filled with GLOBAL_LOAD_ASYNC_TO_LDS_B128 when available (ASYNCcnt + s_wait_asynccnt),
// zero-filled via DS stores for padded/out-of-range columns, shared by all 4 waves.
// mode 0: 3x3 "same" conv (pad 1).
// mode 1: stride-2 transposed conv parity decomposition: blockIdx.z = batch*4 + (py*2+px);
//         tap (ty,tx) active iff ty&1==py && tx&1==px, input offset (-(ty>>1), -(tx>>1)),
//         out pixel (2a+py, 2b+px) of the (2H+1)x(2W+1) plane.
__global__ void __launch_bounds__(128)
conv_wmma(const unsigned short* __restrict__ xin,   // NHWC bf16 [B][Hin][Win][Cin]
          const unsigned short* __restrict__ wpk,   // [9][Cout][Cin] bf16
          const float* __restrict__ bias,           // [Cout] or nullptr
          float* __restrict__ out,                  // NCHW f32 [B][Cout][OutH][OutW]
          int Hin, int Win, int Cin, int Cout,
          int OutH, int OutW, int mode, int npar)
{
    __shared__ __align__(16) unsigned Als[64 * 16];   // [m][k-dword]: 64 x 32 bf16
    __shared__ __align__(16) unsigned Bls[64 * 16];   // [n][k-dword]

    const int par   = (int)blockIdx.z % npar;
    const int batch = (int)blockIdx.z / npar;
    const int py = par >> 1, px = par & 1;
    const int An = (mode == 0) ? Hin : (Hin + 1 - py);
    const int Bn = (mode == 0) ? Win : (Win + 1 - px);
    const int N  = An * Bn;

    const int tid  = threadIdx.x;
    const int lane = tid & 31;          // wave32
    const int wave = tid >> 5;
    const int half = lane >> 4;
    const int l16  = lane & 15;
    const int wm   = wave & 1;          // wave's M half (32 rows)
    const int wn   = wave >> 1;         // wave's N half (32 cols)

    const int Mbase = blockIdx.y * 64;
    const int Nbase = blockIdx.x * 64;

    // staging role: thread stages row/col (tid>>1), dword offset (tid&1)*8 of 16
    const int  srow = tid >> 1;
    const int  sk0  = (tid & 1) * 8;

    // geometry of the staged B column
    const int  nbS    = Nbase + srow;
    const bool svalid = nbS < N;
    const int  sa = svalid ? nbS / Bn : 0;
    const int  sb = svalid ? nbS % Bn : 0;

    // geometry of this lane's two output-column groups
    bool nval[2]; int aO[2], bO[2];
#pragma unroll
    for (int ni = 0; ni < 2; ++ni) {
        const int n = Nbase + wn * 32 + ni * 16 + l16;
        nval[ni] = n < N;
        aO[ni] = nval[ni] ? n / Bn : 0;
        bO[ni] = nval[ni] ? n % Bn : 0;
    }

    const unsigned short* xb = xin + (size_t)batch * Hin * Win * Cin;

    v8f acc00 = {}, acc01 = {}, acc10 = {}, acc11 = {};

    for (int t = 0; t < 9; ++t) {
        const int ty = t / 3, tx = t % 3;
        int dy, dx;
        if (mode == 0) { dy = ty - 1; dx = tx - 1; }
        else {
            if (((ty & 1) != py) || ((tx & 1) != px)) continue;   // block-uniform
            dy = -(ty >> 1); dx = -(tx >> 1);
        }
        const int  siy  = sa + dy, six = sb + dx;
        const bool sinb = svalid && (siy >= 0) && (siy < Hin) && (six >= 0) && (six < Win);
        const unsigned* gB = (const unsigned*)(xb + (long long)(siy * Win + six) * Cin);
        const unsigned* gA = (const unsigned*)(wpk + ((size_t)t * Cout + (Mbase + srow)) * Cin);

        for (int k0 = 0; k0 < Cin; k0 += 32) {
            __syncthreads();           // protect LDS WAR from previous step's reads
            const int gb  = (k0 >> 1) + sk0;
            const int lds = srow * 16 + sk0;
#if USE_ASYNC_LDS
            // A tile: two async 16B copies per thread (no VGPR round-trip)
            __builtin_amdgcn_global_load_async_to_lds_b128(
                (gv4i_p)(gA + gb),     (lv4i_p)&Als[lds],     0, 0);
            __builtin_amdgcn_global_load_async_to_lds_b128(
                (gv4i_p)(gA + gb + 4), (lv4i_p)&Als[lds + 4], 0, 0);
            if (sinb) {
                __builtin_amdgcn_global_load_async_to_lds_b128(
                    (gv4i_p)(gB + gb),     (lv4i_p)&Bls[lds],     0, 0);
                __builtin_amdgcn_global_load_async_to_lds_b128(
                    (gv4i_p)(gB + gb + 4), (lv4i_p)&Bls[lds + 4], 0, 0);
            } else {
#pragma unroll
                for (int j = 0; j < 8; ++j) Bls[lds + j] = 0u;    // zero-pad column
            }
#else
#pragma unroll
            for (int j = 0; j < 8; ++j)
                Als[lds + j] = gA[gb + j];
            if (sinb) {
#pragma unroll
                for (int j = 0; j < 8; ++j)
                    Bls[lds + j] = gB[gb + j];
            } else {
#pragma unroll
                for (int j = 0; j < 8; ++j)
                    Bls[lds + j] = 0u;
            }
#endif
            if (k0 + 32 < Cin) {
                __builtin_prefetch(gA + gb + 16, 0, 1);            // next K block
                if (sinb) __builtin_prefetch(gB + gb + 16, 0, 1);
            }
#if USE_ASYNC_LDS
            __builtin_amdgcn_s_wait_asynccnt(0);                   // asyncs landed in LDS
#endif
            __syncthreads();

            // fragment loads from LDS
            FragBF a0, a1, b0, b1;
#pragma unroll
            for (int r = 0; r < 8; ++r) {
                // A 16x32 bf16 layout: K pair base (r<4 ? 2r : 2r+8) + 8*half -> dword index
                const int ka = ((r < 4) ? r : (r + 4)) + 4 * half;
                a0.u[r] = Als[(wm * 32 +      l16) * 16 + ka];
                a1.u[r] = Als[(wm * 32 + 16 + l16) * 16 + ka];
                // B 32x16 bf16 layout: K pair base (r<4 ? 2r : 2r+8) + 8*half -> dword index
                const int kb = ((r < 4) ? r : (r + 4)) + 4 * half;
                b0.u[r] = Bls[(wn * 32 +      l16) * 16 + kb];
                b1.u[r] = Bls[(wn * 32 + 16 + l16) * 16 + kb];
            }
            acc00 = __builtin_amdgcn_wmma_f32_16x16x32_bf16(false, a0.v, false, b0.v, (short)0, acc00, false, false);
            acc01 = __builtin_amdgcn_wmma_f32_16x16x32_bf16(false, a0.v, false, b1.v, (short)0, acc01, false, false);
            acc10 = __builtin_amdgcn_wmma_f32_16x16x32_bf16(false, a1.v, false, b0.v, (short)0, acc10, false, false);
            acc11 = __builtin_amdgcn_wmma_f32_16x16x32_bf16(false, a1.v, false, b1.v, (short)0, acc11, false, false);
        }
    }

    float* obase = out + (size_t)batch * Cout * OutH * OutW;
    const int co0 = Mbase + wm * 32;
    epilog_tile(acc00, nval[0], aO[0], bO[0], mode, py, px, obase, OutH, OutW, co0,      half, bias);
    epilog_tile(acc01, nval[1], aO[1], bO[1], mode, py, px, obase, OutH, OutW, co0,      half, bias);
    epilog_tile(acc10, nval[0], aO[0], bO[0], mode, py, px, obase, OutH, OutW, co0 + 16, half, bias);
    epilog_tile(acc11, nval[1], aO[1], bO[1], mode, py, px, obase, OutH, OutW, co0 + 16, half, bias);
}

// ---------------- 5) InstanceNorm apply + modulate -> NHWC bf16 ----------------
__global__ void modulate_pack(const float* __restrict__ x,
                              const float* __restrict__ gamma,
                              const float* __restrict__ beta,
                              const float* __restrict__ mu,
                              const float* __restrict__ rs,
                              unsigned short* __restrict__ xm,
                              int C, int HW, int total)
{
    int idx = blockIdx.x * blockDim.x + threadIdx.x;
    if (idx >= total) return;
    int p  = idx % HW;
    int bc = idx / HW;
    int c  = bc % C;
    int b  = bc / C;
    float xn = (x[idx] - mu[bc]) * rs[bc];
    float v  = xn * gamma[idx] + beta[idx];
    xm[((size_t)b * HW + p) * C + c] = f32_to_bf16(v);
}

// ---------------- 6) Depthwise 4x4 blur + bias + LeakyReLU*sqrt(2) ----------------
__global__ void blur_bias_act(const float* __restrict__ u, const float* __restrict__ abias,
                              float* __restrict__ out, int C, int OH, int UH, int total)
{
    int idx = blockIdx.x * blockDim.x + threadIdx.x;
    if (idx >= total) return;
    const float K4[4] = {0.25f, 0.75f, 0.75f, 0.25f};  // bk = (k/4) outer (k/4)
    int xo = idx % OH; int r = idx / OH;
    int yo = r % OH;   r /= OH;
    int c  = r % C;    int b = r / C;
    const float* up = u + (size_t)(b * C + c) * UH * UH;
    float s = 0.f;
#pragma unroll
    for (int dy = 0; dy < 4; ++dy) {
        int iy = yo - 1 + dy;
        if ((unsigned)iy >= (unsigned)UH) continue;
        const float* row = up + (size_t)iy * UH;
        float ky = K4[dy];
#pragma unroll
        for (int dx = 0; dx < 4; ++dx) {
            int ix = xo - 1 + dx;
            if ((unsigned)ix >= (unsigned)UH) continue;
            s += row[ix] * (ky * K4[dx]);
        }
    }
    float v = s + abias[c];
    out[idx] = (v > 0.f ? v : 0.2f * v) * 1.41421356237f;
}

// ---------------- launcher ----------------
extern "C" void kernel_launch(void* const* d_in, const int* in_sizes, int n_in,
                              void* d_out, int out_size, void* d_ws, size_t ws_size,
                              hipStream_t stream)
{
    (void)in_sizes; (void)n_in; (void)out_size; (void)ws_size;
    const float* x       = (const float*)d_in[0];
    const float* style   = (const float*)d_in[1];
    const float* weight  = (const float*)d_in[2];
    const float* gamma_w = (const float*)d_in[3];
    const float* gamma_b = (const float*)d_in[4];
    const float* beta_w  = (const float*)d_in[5];
    const float* beta_b  = (const float*)d_in[6];
    const float* act_b   = (const float*)d_in[7];
    float* out = (float*)d_out;

    const int B = 4, C = 512, H = 64, W = 64, HW = H * W;
    const int UH = 2 * H + 1;  // 129

    char* ws = (char*)d_ws;
    size_t off = 0;
    auto alloc = [&](size_t bytes) -> char* {
        char* p = ws + off;
        off += (bytes + 255) & ~(size_t)255;
        return p;
    };

    float*          mu = (float*)alloc((size_t)B * C * sizeof(float));
    float*          rs = (float*)alloc((size_t)B * C * sizeof(float));
    unsigned short* wg = (unsigned short*)alloc((size_t)9 * C * C * 2);
    unsigned short* wb = (unsigned short*)alloc((size_t)9 * C * C * 2);
    unsigned short* wm = (unsigned short*)alloc((size_t)9 * C * C * 2);
    unsigned short* xm = (unsigned short*)alloc((size_t)B * HW * C * 2);

    // overlay region: [style_p | gamma | beta] first, then reused as u (129x129 f32)
    char* rb = ws + off;
    unsigned short* sp  = (unsigned short*)rb;
    float*          gws = (float*)(rb + (size_t)B * HW * C * 2);
    float*          bws = (float*)(rb + (size_t)B * HW * C * 2 + (size_t)B * C * HW * 4);
    float*          u   = (float*)rb;

    // 1) stats
    instnorm_stats<<<B * C, 256, 0, stream>>>(x, mu, rs, HW);

    // 2) pack weights (EqualConv scale 1/sqrt(Cin*9) for all three)
    const float wsc = 1.0f / sqrtf((float)(C * 9));
    const int wtotal = C * C * 9;
    pack_weight<<<(wtotal + 255) / 256, 256, 0, stream>>>(gamma_w, wg, wsc, C, C, wtotal);
    pack_weight<<<(wtotal + 255) / 256, 256, 0, stream>>>(beta_w,  wb, wsc, C, C, wtotal);
    pack_weight<<<(wtotal + 255) / 256, 256, 0, stream>>>(weight,  wm, wsc, C, C, wtotal);

    // 3) pack style NHWC bf16
    const int atotal = B * C * HW;
    pack_nhwc<<<(atotal + 255) / 256, 256, 0, stream>>>(style, sp, C, HW, atotal);

    // 4) gamma / beta 3x3 convs (WMMA implicit GEMM, 64x64 macro-tiles)
    dim3 blk(128, 1, 1);
    dim3 g1((HW + 63) / 64, C / 64, B);          // 64 x 8 x 4
    conv_wmma<<<g1, blk, 0, stream>>>(sp, wg, gamma_b, gws, H, W, C, C, H, W, 0, 1);
    conv_wmma<<<g1, blk, 0, stream>>>(sp, wb, beta_b,  bws, H, W, C, C, H, W, 0, 1);

    // 5) instance-norm + modulate -> bf16 NHWC
    modulate_pack<<<(atotal + 255) / 256, 256, 0, stream>>>(x, gws, bws, mu, rs, xm, C, HW, atotal);

    // 6) stride-2 transposed conv, 4 parities per batch -> u[B][C][129][129]
    const int Nmax = (H + 1) * (W + 1);          // 4225
    dim3 g2((Nmax + 63) / 64, C / 64, B * 4);    // 67 x 8 x 16
    conv_wmma<<<g2, blk, 0, stream>>>(xm, wm, nullptr, u, H, W, C, C, UH, UH, 1, 4);

    // 7) blur + bias + leaky*sqrt(2)
    const int ototal = B * C * (2 * H) * (2 * W);
    blur_bias_act<<<(ototal + 255) / 256, 256, 0, stream>>>(u, act_b, out, C, 2 * H, UH, ototal);
}
